// MagnetAttention_78013785964751
// MI455X (gfx1250) — matrix operand order repaired and checked by reading, not compiled, not run on hardware
//
#include <hip/hip_runtime.h>

// MHA block for B=2, S=2048, D=1024, H=16, HD=64 on gfx1250.
// bf16 WMMA (v_wmma_f32_16x16x32_bf16) everywhere; fp32 accumulate.
// Softmax row reductions via DPP16 butterflies (VALU-only, no LDS pipe).

#define BATCH 2
#define SEQ   2048
#define DIM   1024
#define NH    16
#define HDIM  64
#define SCALE 0.125f   // HD^-0.5 = 1/8

typedef __attribute__((ext_vector_type(16))) __bf16 v16bf;
typedef __attribute__((ext_vector_type(8)))  __bf16 v8bf;
typedef __attribute__((ext_vector_type(8)))  float  v8f;

// ---- fragment helpers -------------------------------------------------------
// 16-bit A-matrix 16x32 layout (contiguous K order): lane holds row m=lane&15,
// K {c0..c0+15} with c0 = (lane>=16 ? 16 : 0)+kbase.
// -> two 16B loads at p and p+8.
static __device__ inline v16bf load_a_frag(const __bf16* p) {
  v8bf lo = *(const v8bf*)p;
  v8bf hi = *(const v8bf*)(p + 8);
  v16bf r;
#pragma unroll
  for (int i = 0; i < 8; ++i) { r[i] = lo[i]; r[i + 8] = hi[i]; }
  return r;
}

// ---- DPP16 cross-lane reductions -------------------------------------------
// C-matrix rows live across 16 lanes (one half-wave). Butterfly with DPP:
//   step1: quad_perm(1,0,3,2)  == lane^1
//   step2: quad_perm(2,3,0,1)  == lane^2
//   step3: ROW_HALF_MIRROR     == lane^7  (equivalent to ^4 after quad reduce)
//   step4: ROW_MIRROR          == lane^15 (equivalent to ^8 after 8-reduce)
#define DPP_QUAD_XOR1  0xB1   // perm(1,0,3,2)
#define DPP_QUAD_XOR2  0x4E   // perm(2,3,0,1)
#define DPP_ROW_HMIRR  0x141
#define DPP_ROW_MIRR   0x140

template <int CTRL>
static __device__ inline float dpp_movf(float x) {
  int i = __builtin_bit_cast(int, x);
  i = __builtin_amdgcn_mov_dpp(i, CTRL, 0xF, 0xF, true);
  return __builtin_bit_cast(float, i);
}

static __device__ inline float rowmax16(float v) {
  v = fmaxf(v, dpp_movf<DPP_QUAD_XOR1>(v));
  v = fmaxf(v, dpp_movf<DPP_QUAD_XOR2>(v));
  v = fmaxf(v, dpp_movf<DPP_ROW_HMIRR>(v));
  v = fmaxf(v, dpp_movf<DPP_ROW_MIRR>(v));
  return v;
}
static __device__ inline float rowsum16(float v) {
  v += dpp_movf<DPP_QUAD_XOR1>(v);
  v += dpp_movf<DPP_QUAD_XOR2>(v);
  v += dpp_movf<DPP_ROW_HMIRR>(v);
  v += dpp_movf<DPP_ROW_MIRR>(v);
  return v;
}

// ---- prepass: fp32 -> bf16 --------------------------------------------------
__global__ __launch_bounds__(256) void cvt_bf16_kernel(
    const float* __restrict__ in, __bf16* __restrict__ out, int n) {
  int i = blockIdx.x * blockDim.x + threadIdx.x;
  for (; i < n; i += gridDim.x * blockDim.x) out[i] = (__bf16)in[i];
}

// W[k][n] (DxD row-major) -> Wt[n][k] bf16, LDS-tiled transpose.
__global__ __launch_bounds__(256) void transpose_cvt_kernel(
    const float* __restrict__ W, __bf16* __restrict__ Wt) {
  __shared__ float tile[32][33];
  const int k0 = blockIdx.x * 32, n0 = blockIdx.y * 32;
  for (int i = threadIdx.y; i < 32; i += 8)
    tile[i][threadIdx.x] = W[(size_t)(k0 + i) * DIM + n0 + threadIdx.x];
  __syncthreads();
  for (int i = threadIdx.y; i < 32; i += 8)
    Wt[(size_t)(n0 + i) * DIM + k0 + threadIdx.x] = (__bf16)tile[threadIdx.x][i];
}

// ---- GEMM: C[M=4096,N=1024] = A(bf16) x Wt^T + bias -------------------------
// 256 threads = 8 waves; wave computes 16(M) x 64(N); block tile 128x64.
// MODE 0: Q  -> bf16 [B,H,S,HD], *SCALE
// MODE 1: K  -> bf16 [B,H,S,HD]
// MODE 2: V  -> bf16 [B,H,HD,S]  (transposed for P*V B-fragments)
// MODE 4: final -> fp32 [M,N]
template <int MODE>
__global__ __launch_bounds__(256) void gemm_kernel(
    const __bf16* __restrict__ A, const __bf16* __restrict__ Wt,
    const float* __restrict__ bias, __bf16* __restrict__ outb,
    float* __restrict__ outf) {
  const int wave = threadIdx.x >> 5;
  const int lane = threadIdx.x & 31;
  const int col  = lane & 15;
  const int half = lane >> 4;
  const int rowBase = blockIdx.x * 128 + wave * 16;
  const int colBase = blockIdx.y * 64;

  const __bf16* aBase = A + (size_t)(rowBase + col) * DIM + half * 16;
  v8f acc[4] = {};

  for (int kb = 0; kb < DIM; kb += 32) {
    v16bf af = load_a_frag(aBase + kb);
#pragma unroll
    for (int nt = 0; nt < 4; ++nt) {
      const __bf16* bp =
          Wt + (size_t)(colBase + nt * 16 + col) * DIM + kb + half * 16;
      v16bf bf = *(const v16bf*)bp;  // 32B aligned contiguous K chunk
      acc[nt] = __builtin_amdgcn_wmma_f32_16x16x32_bf16(
          false, af, false, bf, (short)0, acc[nt], false, false);
    }
  }

#pragma unroll
  for (int nt = 0; nt < 4; ++nt) {
    const int n = colBase + nt * 16 + col;
    const float bv = bias[n];
#pragma unroll
    for (int j = 0; j < 8; ++j) {
      const int m = rowBase + j + half * 8;  // C layout: vgpr j -> row j/(j+8)
      float v = acc[nt][j] + bv;
      if (MODE == 0) v *= SCALE;
      if (MODE == 4) {
        outf[(size_t)m * DIM + n] = v;
      } else {
        const int b = m >> 11, s = m & (SEQ - 1);
        const int h = n >> 6, hd = n & (HDIM - 1);
        size_t idx;
        if (MODE == 2)
          idx = (((size_t)b * NH + h) * HDIM + hd) * SEQ + s;   // V^T
        else
          idx = (((size_t)b * NH + h) * SEQ + s) * HDIM + hd;   // Q/K
        outb[idx] = (__bf16)v;
      }
    }
  }
}

// ---- flash attention: block = (b, h, 128 q-rows); wave = 16 q-rows ----------
__global__ __launch_bounds__(256) void attn_kernel(
    const __bf16* __restrict__ Q, const __bf16* __restrict__ K,
    const __bf16* __restrict__ VT, const int* __restrict__ qmask,
    const int* __restrict__ kmask, __bf16* __restrict__ Ob) {
  __shared__ __align__(16) __bf16 pbuf[8][16 * 32];  // per-wave P tile

  const int wave = threadIdx.x >> 5;
  const int lane = threadIdx.x & 31;
  const int col  = lane & 15;
  const int half = lane >> 4;
  const int h = blockIdx.y, b = blockIdx.z;
  const int qBase = blockIdx.x * 128 + wave * 16;

  const size_t bh = (size_t)b * NH + h;
  const __bf16* Qh = Q  + bh * SEQ * HDIM;
  const __bf16* Kh = K  + bh * SEQ * HDIM;
  const __bf16* Vh = VT + bh * HDIM * SEQ;
  const int* km = kmask + b * SEQ;
  const int* qm = qmask + b * SEQ;

  // Q A-fragments for this wave's 16 rows (K chunks 0-31 and 32-63), loaded once.
  const __bf16* qrow = Qh + (size_t)(qBase + col) * HDIM;
  const v16bf aq0 = load_a_frag(qrow + half * 16);
  const v16bf aq1 = load_a_frag(qrow + 32 + half * 16);

  v8f acc[4] = {};
  float mrow[8], lrow[8];
#pragma unroll
  for (int j = 0; j < 8; ++j) { mrow[j] = -1e30f; lrow[j] = 0.f; }

  for (int kb = 0; kb < SEQ; kb += 32) {
    if (km[kb] == 0) break;  // padding is contiguous at the end
    if (kb + 32 < SEQ)
      __builtin_prefetch(Kh + (size_t)(kb + 32 + col) * HDIM, 0, 1);

    // scores: two 16x16 tiles (keys kb.. and kb+16..), each = 2 WMMAs over HD=64
    v8f s0 = {}, s1 = {};
    {
      const __bf16* k0 = Kh + (size_t)(kb + col) * HDIM + half * 16;
      v16bf b0 = *(const v16bf*)k0;
      v16bf b1 = *(const v16bf*)(k0 + 32);
      s0 = __builtin_amdgcn_wmma_f32_16x16x32_bf16(false, aq0, false, b0, (short)0, s0, false, false);
      s0 = __builtin_amdgcn_wmma_f32_16x16x32_bf16(false, aq1, false, b1, (short)0, s0, false, false);
    }
    {
      const __bf16* k1 = Kh + (size_t)(kb + 16 + col) * HDIM + half * 16;
      v16bf b0 = *(const v16bf*)k1;
      v16bf b1 = *(const v16bf*)(k1 + 32);
      s1 = __builtin_amdgcn_wmma_f32_16x16x32_bf16(false, aq0, false, b0, (short)0, s1, false, false);
      s1 = __builtin_amdgcn_wmma_f32_16x16x32_bf16(false, aq1, false, b1, (short)0, s1, false, false);
    }

    const float kb0 = km[kb + col]      ? 0.f : -1e30f;  // per-column key mask
    const float kb1 = km[kb + 16 + col] ? 0.f : -1e30f;

    // online softmax per row (row = j + 8*half, shared across the 16 lanes)
#pragma unroll
    for (int j = 0; j < 8; ++j) {
      const float a0 = s0[j] + kb0;
      const float a1 = s1[j] + kb1;
      const float tm = rowmax16(fmaxf(a0, a1));
      const float mn = fmaxf(mrow[j], tm);
      const float sc = __expf(mrow[j] - mn);
      const float p0 = __expf(a0 - mn);
      const float p1 = __expf(a1 - mn);
      lrow[j] = lrow[j] * sc + rowsum16(p0 + p1);
      mrow[j] = mn;
#pragma unroll
      for (int nt = 0; nt < 4; ++nt) acc[nt][j] *= sc;
      pbuf[wave][(j + half * 8) * 32 + col]      = (__bf16)p0;  // C->LDS
      pbuf[wave][(j + half * 8) * 32 + 16 + col] = (__bf16)p1;
    }

    // reload P as an A fragment (16 rows x 32 keys) and do P*V (HD=64 -> 4 tiles)
    v16bf pa = load_a_frag(&pbuf[wave][col * 32 + half * 16]);
#pragma unroll
    for (int nt = 0; nt < 4; ++nt) {
      const __bf16* vp = Vh + (size_t)(nt * 16 + col) * SEQ + kb + half * 16;
      v16bf bv = *(const v16bf*)vp;
      acc[nt] = __builtin_amdgcn_wmma_f32_16x16x32_bf16(
          false, pa, false, bv, (short)0, acc[nt], false, false);
    }
  }

  // epilogue: normalize, zero padded-query rows, store bf16 [B,S,D] for Wo GEMM
#pragma unroll
  for (int j = 0; j < 8; ++j) {
    const int s = qBase + j + half * 8;
    const float inv = (qm[s] != 0 && lrow[j] > 0.f) ? 1.0f / lrow[j] : 0.f;
#pragma unroll
    for (int nt = 0; nt < 4; ++nt) {
      Ob[((size_t)b * SEQ + s) * DIM + h * HDIM + nt * 16 + col] =
          (__bf16)(acc[nt][j] * inv);
    }
  }
}

// ---- launch -----------------------------------------------------------------
extern "C" void kernel_launch(void* const* d_in, const int* in_sizes, int n_in,
                              void* d_out, int out_size, void* d_ws, size_t ws_size,
                              hipStream_t stream) {
  const float* hs = (const float*)d_in[0];
  const float* Wq = (const float*)d_in[1];
  const float* bq = (const float*)d_in[2];
  const float* Wk = (const float*)d_in[3];
  const float* bk = (const float*)d_in[4];
  const float* Wv = (const float*)d_in[5];
  const float* bv = (const float*)d_in[6];
  const float* Wo = (const float*)d_in[7];
  const float* bo = (const float*)d_in[8];
  const int* qmask = (const int*)d_in[9];
  const int* kmask = (const int*)d_in[10];
  float* out = (float*)d_out;

  // workspace carve-up (bf16 elements), total 24M elems = 48 MB
  const size_t M = (size_t)BATCH * SEQ;        // 4096
  const size_t MD = M * DIM;                   // 4M
  const size_t DD = (size_t)DIM * DIM;         // 1M
  __bf16* ws   = (__bf16*)d_ws;
  __bf16* hsb  = ws;
  __bf16* wtq  = hsb + MD;
  __bf16* wtk  = wtq + DD;
  __bf16* wtv  = wtk + DD;
  __bf16* wto  = wtv + DD;
  __bf16* qb   = wto + DD;
  __bf16* kbuf = qb + MD;
  __bf16* vtb  = kbuf + MD;
  __bf16* attb = vtb + MD;

  // 1) bf16 conversion of activations + transposed bf16 weights
  cvt_bf16_kernel<<<512, 256, 0, stream>>>(hs, hsb, (int)MD);
  dim3 tb(32, 8), tg(DIM / 32, DIM / 32);
  transpose_cvt_kernel<<<tg, tb, 0, stream>>>(Wq, wtq);
  transpose_cvt_kernel<<<tg, tb, 0, stream>>>(Wk, wtk);
  transpose_cvt_kernel<<<tg, tb, 0, stream>>>(Wv, wtv);
  transpose_cvt_kernel<<<tg, tb, 0, stream>>>(Wo, wto);

  // 2) Q/K/V projections (Q scaled; V stored transposed per head)
  dim3 gg((unsigned)(M / 128), DIM / 64);
  gemm_kernel<0><<<gg, 256, 0, stream>>>(hsb, wtq, bq, qb,   nullptr);
  gemm_kernel<1><<<gg, 256, 0, stream>>>(hsb, wtk, bk, kbuf, nullptr);
  gemm_kernel<2><<<gg, 256, 0, stream>>>(hsb, wtv, bv, vtb,  nullptr);

  // 3) flash attention
  attn_kernel<<<dim3(SEQ / 128, NH, BATCH), 256, 0, stream>>>(
      qb, kbuf, vtb, qmask, kmask, attb);

  // 4) output projection -> fp32 d_out
  gemm_kernel<4><<<gg, 256, 0, stream>>>(attb, wto, bo, nullptr, out);

  (void)in_sizes; (void)n_in; (void)out_size; (void)ws_size;
}
